// NetVLAD_58540404244567
// MI455X (gfx1250) — matrix operand order repaired and checked by reading, not compiled, not run on hardware
//
#include <hip/hip_runtime.h>
#include <cstdint>

#define BDIM   64
#define LDIM   4096
#define DDIM   128
#define KDIM   64
#define LCHUNK 256
#define NSPLIT (LDIM / LCHUNK)   // 16 chunks per batch row-dim

typedef __attribute__((ext_vector_type(16))) __bf16 v16bf;
typedef __attribute__((ext_vector_type(8)))  float  v8f;

union FragBF { v16bf v; uint4 q[2]; };

__device__ __forceinline__ unsigned short f2bf(float f) {
  unsigned int u = __float_as_uint(f);
  u += 0x7FFFu + ((u >> 16) & 1u);           // round-to-nearest-even
  return (unsigned short)(u >> 16);
}

// CDNA5 async global->LDS copy, 16B per lane, ASYNCcnt-tracked (no VGPR data).
__device__ __forceinline__ void async_load_b128(const void* gptr, const void* lds_ptr) {
  const unsigned lds_off = (unsigned)(uintptr_t)lds_ptr;              // flat LDS addr[31:0]
  const unsigned long long ga = (unsigned long long)(uintptr_t)gptr;
  asm volatile("global_load_async_to_lds_b128 %0, %1, off"
               :: "v"(lds_off), "v"(ga) : "memory");
}

__device__ __forceinline__ void wait_async_zero() {
#if __has_builtin(__builtin_amdgcn_s_wait_asynccnt)
  __builtin_amdgcn_s_wait_asynccnt(0);
#else
  asm volatile("s_wait_asynccnt 0x0" ::: "memory");
#endif
}

// ---------------------------------------------------------------- kernel Z
__global__ void zero_ws(float* __restrict__ p, int n) {
  int i = blockIdx.x * blockDim.x + threadIdx.x;
  if (i < n) p[i] = 0.0f;
}

// ---------------------------------------------------------------- kernel A
// Per-descriptor L2 normalization of x, stored as raw-bf16. One wave per row.
__global__ void normalize_rows(const float* __restrict__ x,
                               unsigned short* __restrict__ xn) {
  const int lane = threadIdx.x & 31;
  const int wave = threadIdx.x >> 5;
  const size_t row = (size_t)blockIdx.x * 8 + wave;
  const float4 v = ((const float4*)(x + row * DDIM))[lane];
  float ss = v.x * v.x + v.y * v.y + v.z * v.z + v.w * v.w;
#pragma unroll
  for (int off = 16; off; off >>= 1) ss += __shfl_xor(ss, off, 32);
  const float sc = 1.0f / fmaxf(sqrtf(ss), 1e-12f);
  const unsigned int lo = (unsigned)f2bf(v.x * sc) | ((unsigned)f2bf(v.y * sc) << 16);
  const unsigned int hi = (unsigned)f2bf(v.z * sc) | ((unsigned)f2bf(v.w * sc) << 16);
  ((uint2*)(xn + row * DDIM))[lane] = make_uint2(lo, hi);
}

// ---------------------------------------------------------------- kernel B
// Fused: logits (WMMA) -> softmax (in accumulator layout) -> vlad pooling (WMMA).
// grid.x = BDIM * NSPLIT, block = 256 (8 wave32).
// LDS: w (16.25KB) | xn row-major (64KB) | xn d-major (64KB) | a^T (32KB)
#define OFF_BIAS 16384
#define OFF_XROW 16640
#define OFF_XNT  (OFF_XROW + LCHUNK * DDIM * 2)   // 82176
#define OFF_AT   (OFF_XNT  + LCHUNK * DDIM * 2)   // 147712
#define SMEM_BYTES (OFF_AT + KDIM * LCHUNK * 2)   // 180480

__global__ __launch_bounds__(256, 1)
void netvlad_fused(const unsigned short* __restrict__ xn,
                   const float* __restrict__ assign_w,
                   const float* __restrict__ assign_b,
                   float* __restrict__ vlad_acc,
                   float* __restrict__ asum_acc) {
  extern __shared__ char smem[];
  unsigned short* w_lds    = (unsigned short*)smem;                 // [64][128] bf16
  float*          bias_lds = (float*)(smem + OFF_BIAS);             // [64]
  unsigned short* xrow_lds = (unsigned short*)(smem + OFF_XROW);    // [256][128] bf16
  unsigned short* xnT_lds  = (unsigned short*)(smem + OFF_XNT);     // [128][256] bf16
  unsigned short* at_lds   = (unsigned short*)(smem + OFF_AT);      // [64][256]  bf16

  const int tid  = threadIdx.x;
  const int lane = tid & 31;
  const int wave = tid >> 5;
  const int b     = blockIdx.x / NSPLIT;
  const int chunk = blockIdx.x % NSPLIT;
  const int l0    = chunk * LCHUNK;

  const v8f vzero = {0.f, 0.f, 0.f, 0.f, 0.f, 0.f, 0.f, 0.f};

  // ---- stage 1: async DMA of the bf16 xn chunk into LDS (row-major) -----
  const unsigned short* xchunk = xn + ((size_t)b * LDIM + l0) * DDIM;
#pragma unroll
  for (int i = tid; i < (LCHUNK * DDIM) / 8; i += 256)   // 16B per lane/iter
    async_load_b128(xchunk + i * 8, xrow_lds + i * 8);

  // weights + bias staged through VGPRs (need f32 -> bf16 conversion)
  for (int i = tid; i < KDIM * DDIM; i += 256) w_lds[i] = f2bf(assign_w[i]);
  if (tid < KDIM) bias_lds[tid] = assign_b[tid];

  wait_async_zero();
  __syncthreads();

  // ---- stage 2: LDS->LDS transpose to d-major for pass-2 B fragments ----
  for (int i = tid; i < (LCHUNK * DDIM) / 8; i += 256) {
    uint4 u = ((const uint4*)xrow_lds)[i];
    const unsigned short* e = (const unsigned short*)&u;
    const int row = (i * 8) >> 7;       // / DDIM
    const int d0  = (i * 8) & 127;
#pragma unroll
    for (int j = 0; j < 8; ++j) xnT_lds[(d0 + j) * LCHUNK + row] = e[j];
  }
  __syncthreads();

  const int m  = lane & 15;     // row/col index within fragment halves
  const int hs = lane >> 4;     // which 16-lane half

  // ---- pass 1: logits -> softmax -> a^T (32 rows per wave) --------------
  float asum_part[4] = {0.f, 0.f, 0.f, 0.f};
  for (int h = 0; h < 2; ++h) {
    const int r0 = wave * 32 + h * 16;                  // row tile within chunk
    v8f acc[4] = {vzero, vzero, vzero, vzero};
    const unsigned short* rowp = xrow_lds + (r0 + m) * DDIM;  // A row (M = lane&15)
#pragma unroll
    for (int kk = 0; kk < 4; ++kk) {                    // K(=D) in steps of 32
      const int d0 = kk * 32;
      FragBF A;
      A.q[0] = *(const uint4*)(rowp + d0 + hs * 8);
      A.q[1] = *(const uint4*)(rowp + d0 + 16 + hs * 8);
#pragma unroll
      for (int nt = 0; nt < 4; ++nt) {                  // N(=K clusters) tiles
        const unsigned short* wp = w_lds + (nt * 16 + m) * DDIM + d0 + hs * 8;
        FragBF Bm;
        Bm.q[0] = *(const uint4*)(wp);
        Bm.q[1] = *(const uint4*)(wp + 16);
        acc[nt] = __builtin_amdgcn_wmma_f32_16x16x32_bf16(
            false, A.v, false, Bm.v, (short)0, acc[nt], false, false);
      }
    }
    // bias: each lane owns a fixed N column per tile
#pragma unroll
    for (int nt = 0; nt < 4; ++nt) {
      const float bn = bias_lds[nt * 16 + m];
#pragma unroll
      for (int r = 0; r < 8; ++r) acc[nt][r] += bn;
    }
    // softmax over K=64 (4 tiles x 16 lanes) per output row
#pragma unroll
    for (int r = 0; r < 8; ++r) {
      float mx = fmaxf(fmaxf(acc[0][r], acc[1][r]), fmaxf(acc[2][r], acc[3][r]));
#pragma unroll
      for (int off = 8; off; off >>= 1) mx = fmaxf(mx, __shfl_xor(mx, off, 16));
      float e0 = __expf(acc[0][r] - mx);
      float e1 = __expf(acc[1][r] - mx);
      float e2 = __expf(acc[2][r] - mx);
      float e3 = __expf(acc[3][r] - mx);
      float sm = e0 + e1 + e2 + e3;
#pragma unroll
      for (int off = 8; off; off >>= 1) sm += __shfl_xor(sm, off, 16);
      const float inv = 1.0f / sm;
      e0 *= inv; e1 *= inv; e2 *= inv; e3 *= inv;
      asum_part[0] += e0; asum_part[1] += e1;
      asum_part[2] += e2; asum_part[3] += e3;
      const int row = r0 + r + hs * 8;                  // C layout: halves hold M, M+8
      at_lds[(0 * 16 + m) * LCHUNK + row] = f2bf(e0);
      at_lds[(1 * 16 + m) * LCHUNK + row] = f2bf(e1);
      at_lds[(2 * 16 + m) * LCHUNK + row] = f2bf(e2);
      at_lds[(3 * 16 + m) * LCHUNK + row] = f2bf(e3);
    }
  }
  // per-cluster soft-assign sums (combine the two row halves, then atomic)
#pragma unroll
  for (int nt = 0; nt < 4; ++nt) {
    float p = asum_part[nt] + __shfl_xor(asum_part[nt], 16, 32);
    if (lane < 16) atomicAdd(&asum_acc[b * KDIM + nt * 16 + m], p);
  }
  __syncthreads();

  // ---- pass 2: vlad[k,d] += a^T @ xn over all 256 chunk rows ------------
  const int kc = wave >> 1;    // cluster tile (0..3)  -> 16 clusters
  const int dh = wave & 1;     // D half (0..1)        -> 4 d-tiles of 16
  v8f c2[4] = {vzero, vzero, vzero, vzero};
#pragma unroll
  for (int rb = 0; rb < LCHUNK; rb += 32) {
    const unsigned short* ap = at_lds + (kc * 16 + m) * LCHUNK + rb;
    FragBF A;
    A.q[0] = *(const uint4*)(ap + hs * 8);
    A.q[1] = *(const uint4*)(ap + 16 + hs * 8);
#pragma unroll
    for (int j = 0; j < 4; ++j) {
      const int dcol = (dh * 4 + j) * 16 + m;
      const unsigned short* bp = xnT_lds + dcol * LCHUNK + rb + hs * 8;
      FragBF Bm;
      Bm.q[0] = *(const uint4*)(bp);
      Bm.q[1] = *(const uint4*)(bp + 16);
      c2[j] = __builtin_amdgcn_wmma_f32_16x16x32_bf16(
          false, A.v, false, Bm.v, (short)0, c2[j], false, false);
    }
  }
#pragma unroll
  for (int j = 0; j < 4; ++j) {
    const int dcol = (dh * 4 + j) * 16 + m;
#pragma unroll
    for (int r = 0; r < 8; ++r) {
      const int cl = kc * 16 + r + hs * 8;
      atomicAdd(&vlad_acc[((size_t)b * KDIM + cl) * DDIM + dcol], c2[j][r]);
    }
  }
}

// ---------------------------------------------------------------- kernel C
// vlad -= asum*centroid; intra-cluster L2 norm; inference BatchNorm.
__global__ void finalize(const float* __restrict__ vlad_acc,
                         const float* __restrict__ asum_acc,
                         const float* __restrict__ centroids,
                         const float* __restrict__ g,
                         const float* __restrict__ be,
                         const float* __restrict__ mu,
                         const float* __restrict__ var,
                         float* __restrict__ out) {
  __shared__ float sp[4];
  const int bk = blockIdx.x;            // b*64 + k
  const int k  = bk & 63;
  const int d  = threadIdx.x;           // 0..127
  float v = vlad_acc[(size_t)bk * DDIM + d] - asum_acc[bk] * centroids[k * DDIM + d];
  float ss = v * v;
#pragma unroll
  for (int off = 16; off; off >>= 1) ss += __shfl_xor(ss, off, 32);
  if ((threadIdx.x & 31) == 0) sp[threadIdx.x >> 5] = ss;
  __syncthreads();
  const float tot = sp[0] + sp[1] + sp[2] + sp[3];
  const float vn = v / fmaxf(sqrtf(tot), 1e-12f);
  const int gi = k * DDIM + d;
  out[(size_t)bk * DDIM + d] =
      g[gi] * (vn - mu[gi]) * rsqrtf(var[gi] + 1e-5f) + be[gi];
}

// ---------------------------------------------------------------- launch
extern "C" void kernel_launch(void* const* d_in, const int* in_sizes, int n_in,
                              void* d_out, int out_size, void* d_ws, size_t ws_size,
                              hipStream_t stream) {
  (void)in_sizes; (void)n_in; (void)out_size; (void)ws_size;
  const float* x         = (const float*)d_in[0];
  const float* assign_w  = (const float*)d_in[1];
  const float* assign_b  = (const float*)d_in[2];
  const float* centroids = (const float*)d_in[3];
  const float* bn_gamma  = (const float*)d_in[4];
  const float* bn_beta   = (const float*)d_in[5];
  const float* bn_mean   = (const float*)d_in[6];
  const float* bn_var    = (const float*)d_in[7];
  float* out = (float*)d_out;

  // workspace: [ xn bf16 : B*L*D*2 = 64MB ][ vlad_acc f32 2MB ][ asum f32 16KB ]
  unsigned short* xn = (unsigned short*)d_ws;
  float* vlad_acc = (float*)((char*)d_ws + (size_t)BDIM * LDIM * DDIM * 2);
  float* asum_acc = vlad_acc + (size_t)BDIM * KDIM * DDIM;

  const int nzero = BDIM * KDIM * DDIM + BDIM * KDIM;
  zero_ws<<<(nzero + 255) / 256, 256, 0, stream>>>(vlad_acc, nzero);
  normalize_rows<<<(BDIM * LDIM) / 8, 256, 0, stream>>>(x, xn);
  netvlad_fused<<<BDIM * NSPLIT, 256, SMEM_BYTES, stream>>>(
      xn, assign_w, assign_b, vlad_acc, asum_acc);
  finalize<<<BDIM * KDIM, 128, 0, stream>>>(
      vlad_acc, asum_acc, centroids, bn_gamma, bn_beta, bn_mean, bn_var, out);
}
